// EarModel_32658931319637
// MI455X (gfx1250) — matrix operand order repaired and checked by reading, not compiled, not run on hardware
//
#include <hip/hip_runtime.h>
#include <math.h>

#define NSAMP 120000
#define NCH 32
#define FSR 24000.0f
#define SMALLV 1e-30f
#define LAG0 (-2399)
#define NLAG 4800
#define NQBLK 19
#define NLBUF (NQBLK * 256)
#define TWO_PI_D 6.283185307179586
#define PADE 2816                 // f16 zero-pad (elements) each side, multiple of 16
#define PLEN (NSAMP + 2 * PADE)   // 125632 elements per padded stream

typedef __attribute__((ext_vector_type(16))) _Float16 v16h;
typedef __attribute__((ext_vector_type(8)))  _Float16 v8h;
typedef __attribute__((ext_vector_type(8)))  float    v8f;

// ---- parameter block layout (float indices into ws[0..1023]) ----
#define P_CF     0
#define P_BW1    32
#define P_ATTN   64    // [2][32] x then y
#define P_BWMIN  128
#define P_KNEE   192
#define P_CR     256
#define P_IHC    320
#define P_BW     384   // bw_x [32], bw_y [32]
#define P_CTLRMS 448
#define P_ENVRMS 512
#define P_GD     576   // int correct[32]
#define P_DELAY0 608   // int
#define P_DELAYS 640   // int[64]

// ================= device helpers =================
__device__ inline float interp_hl(const float* hl, float f) {
  float a[6] = {250.f, 500.f, 1000.f, 2000.f, 4000.f, 6000.f};
  if (f <= a[0]) return hl[0];
  if (f >= a[5]) return hl[5];
  float r = hl[5];
  for (int i = 0; i < 5; ++i) {
    if (f < a[i+1]) { float w = (f - a[i]) / (a[i+1] - a[i]); r = hl[i] + (hl[i+1]-hl[i])*w; break; }
  }
  return r;
}

struct LossP { float attn_ohc, bw, knee, cr, ihc; };
__device__ inline LossP loss_params(float loss, int i) {
  float cr = 1.25f + 2.25f * (float)i / 31.f;
  float tohc = 1.25f * 70.f * (1.f - 1.f / cr);
  float ao = (loss < tohc) ? 0.8f * loss : 0.8f * tohc;
  float ai = (loss < tohc) ? 0.2f * loss : 0.2f * tohc + loss - tohc;
  float q = ao / 50.f;
  LossP p;
  p.attn_ohc = ao;
  p.bw = 1.f + q + 2.f * q * q * q;
  p.knee = ao + 30.f;
  float up = 30.f + 70.f / cr;
  p.cr = (100.f - p.knee) / (up + ao - p.knee);
  p.ihc = ai;
  return p;
}

struct GT { float a1, a2, a3, a4, a5, gain; double w; };
__device__ inline GT gt_coeffs(float bw, float cf) {
  float tpt = 6.28318530717958647f / FSR;
  float tptbw = tpt * bw * (24.7f + cf / 9.26449f) * 1.019f;
  float a = __expf(-tptbw);
  GT g;
  g.a1 = 4.f * a; g.a2 = -6.f * a * a; g.a3 = 4.f * a * a * a;
  g.a4 = -a * a * a * a; g.a5 = 4.f * a * a;
  g.gain = 2.f * (1.f - g.a1 - g.a2 - g.a3 - g.a4) / (1.f + g.a1 + g.a5);
  g.w = (TWO_PI_D / 24000.0) * (double)cf;
  return g;
}

struct IIR4 { float z0, z1, z2, z3; };
// DF2T, b=[1,a1,a5,0,0], a=[1,-a1,-a2,-a3,-a4]
__device__ inline float iir4_step(IIR4& s, const GT& g, float x) {
  float y = x + s.z0;
  s.z0 = s.z1 + g.a1 * x + g.a1 * y;
  s.z1 = s.z2 + g.a5 * x + g.a2 * y;
  s.z2 = s.z3 + g.a3 * y;
  s.z3 = g.a4 * y;
  return y;
}

__device__ inline unsigned hashu(unsigned x) {
  x ^= x >> 16; x *= 0x7feb352dU; x ^= x >> 15; x *= 0x846ca68bU; x ^= x >> 16; return x;
}
__device__ inline float gaussf(unsigned seed, unsigned idx) {
  unsigned u1 = hashu(seed * 0x9E3779B9u ^ (idx * 2u + 1u));
  unsigned u2 = hashu(seed * 0x85EBCA6Bu ^ (idx * 2u + 2u));
  float f1 = ((float)u1 + 1.f) * (1.f / 4294967296.f);
  float f2 = (float)u2 * (1.f / 4294967296.f);
  float r = sqrtf(-2.f * __logf(f1));
  return r * __cosf(6.2831853f * f2);
}

// ================= kernels =================

__global__ __launch_bounds__(32) void k_params(const float* hl, const int* eqp, float* P) {
  int i = threadIdx.x;
  const double c = 9.26449 * 24.7;
  double cf;
  if (i == 31) cf = 8000.0;
  else {
    double step = (log(80.0 + c) - log(8000.0 + c)) / 31.0;
    cf = -c + exp((double)(31 - i) * step) * (8000.0 + c);
  }
  float cff = (float)cf;
  P[P_CF + i] = cff;
  LossP p1 = loss_params(100.f, i);
  P[P_BW1 + i] = p1.bw;
  float ly = fmaxf(interp_hl(hl, cff), 0.f);
  float lx = (*eqp != 0) ? ly : 0.f;
  LossP px = loss_params(lx, i), py = loss_params(ly, i);
  P[P_ATTN  + i] = px.attn_ohc; P[P_ATTN  + 32 + i] = py.attn_ohc;
  P[P_BWMIN + i] = px.bw;       P[P_BWMIN + 32 + i] = py.bw;
  P[P_KNEE  + i] = px.knee;     P[P_KNEE  + 32 + i] = py.knee;
  P[P_CR    + i] = px.cr;       P[P_CR    + 32 + i] = py.cr;
  P[P_IHC   + i] = px.ihc;      P[P_IHC   + 32 + i] = py.ihc;
}

// f32 stream -> zero-padded f16 stream (pad PADE each side)
__global__ __launch_bounds__(256) void k_tof16(const float* __restrict__ in,
                                               _Float16* __restrict__ out) {
  int s = blockIdx.y;
  int i = blockIdx.x * 256 + threadIdx.x;
  if (i >= PLEN) return;
  int j = i - PADE;
  float v = (j >= 0 && j < NSAMP) ? in[(size_t)s * NSAMP + j] : 0.f;
  out[(size_t)s * PLEN + i] = (_Float16)v;
}

// Cross-correlation as WMMA GEMM: lag = B0 + 16q + r  (B0 = LAG0 + 256*blockIdx.x)
// c[lag] = sum_{t'} A[r,k]*B[k,q], A[r,k]=y[t0+k-B0-r], B[k,q]=x[t0+k+16q]
// x,y point at the signal start inside padded f16 streams (all accesses in-bounds).
__global__ __launch_bounds__(32) void k_corr(const _Float16* __restrict__ xbase,
                                             const _Float16* __restrict__ ybase,
                                             long xstride, long ystride,
                                             float* __restrict__ corr) {
  int qblk = blockIdx.x;
  int pair = blockIdx.y;
  const _Float16* x = xbase + (long)pair * xstride;
  const _Float16* y = ybase + (long)pair * ystride;
  int lane = threadIdx.x;
  int half = lane >> 4;
  int m = lane & 15;
  int B0 = LAG0 + 256 * qblk;
  // per-lane base pointers (loop adds t0)
  const _Float16* ya = y + 16 * half - B0 - m;  // A: element i -> ya[t0+i] (16 contiguous k per half)
  const _Float16* xb = x + 16 * half + 16 * m; // B: 16 contiguous halfs, 32B aligned
  const _Float16* ylim = y + NSAMP + PADE - 16;
  v8f acc = {};
  for (int t0 = -256; t0 < NSAMP; t0 += 32) {
    // prefetch the next tiles (clamped to stay inside the padded buffer)
    const _Float16* pfy = ya + t0 + 2048;
    __builtin_prefetch(pfy < ylim ? pfy : ylim, 0, 1);
    v16h a;
#pragma unroll
    for (int i = 0; i < 8; ++i) {
      a[i]     = ya[t0 + i];
      a[i + 8] = ya[t0 + 8 + i];
    }
    v8h blo = *(const v8h*)(xb + t0);
    v8h bhi = *(const v8h*)(xb + t0 + 8);
    v16h b = __builtin_shufflevector(blo, bhi, 0, 1, 2, 3, 4, 5, 6, 7,
                                     8, 9, 10, 11, 12, 13, 14, 15);
    acc = __builtin_amdgcn_wmma_f32_16x16x32_f16(false, a, false, b, (short)0, acc,
                                                 false, false);
  }
  // D layout: lane N = lane&15, vgpr j -> M = j + 8*half
#pragma unroll
  for (int j = 0; j < 8; ++j) {
    int r = j + 8 * half;
    int idx = 256 * qblk + 16 * m + r;
    corr[(long)pair * NLBUF + idx] = acc[j];
  }
}

__global__ __launch_bounds__(256) void k_argmax(const float* __restrict__ corr,
                                                int* __restrict__ delays) {
  int p = blockIdx.x;
  __shared__ float sv[256];
  __shared__ int   si[256];
  int t = threadIdx.x;
  float best = -3.0e38f; int bi = 0;
  for (int l = t; l < NLAG; l += 256) {
    float v = corr[(long)p * NLBUF + l];
    if (v > best) { best = v; bi = l; }
  }
  sv[t] = best; si[t] = bi;
  __syncthreads();
  for (int s = 128; s > 0; s >>= 1) {
    if (t < s) {
      if (sv[t + s] > sv[t] || (sv[t + s] == sv[t] && si[t + s] < si[t])) {
        sv[t] = sv[t + s]; si[t] = si[t + s];
      }
    }
    __syncthreads();
  }
  if (t == 0) delays[p] = -(LAG0 + si[0]);  // delay = -best_lag
}

__global__ __launch_bounds__(32) void k_midear(const float* __restrict__ ref,
                                               const float* __restrict__ proc,
                                               const float* P, float* mid) {
  int lane = threadIdx.x;
  if (lane >= 2) return;
  int d0 = ((const int*)P)[P_DELAY0];
  float K = __tanf(3.14159265358979f * 5000.f / 24000.f);
  float lb0 = K / (1.f + K), lb1 = lb0, la1 = (K - 1.f) / (K + 1.f);
  float K2 = __tanf(3.14159265358979f * 350.f / 24000.f);
  float nrm = 1.f + 1.41421356f * K2 + K2 * K2;
  float hb0 = 1.f / nrm, hb1 = -2.f / nrm, hb2 = 1.f / nrm;
  float ha1 = 2.f * (K2 * K2 - 1.f) / nrm, ha2 = (1.f - 1.41421356f * K2 + K2 * K2) / nrm;
  float z = 0.f, w0 = 0.f, w1 = 0.f;
  float* out = mid + (size_t)lane * NSAMP;
  for (int t = 0; t < NSAMP; ++t) {
    float x;
    if (lane == 0) x = ref[t];
    else { int j = t + d0; x = (j >= 0 && j < NSAMP) ? proc[j] : 0.f; }
    float y1 = lb0 * x + z; z = lb1 * x - la1 * y1;
    float y2 = hb0 * y1 + w0;
    w0 = w1 + hb1 * y1 - ha1 * y2;
    w1 = hb2 * y1 - ha2 * y2;
    out[t] = y2;
  }
}

__global__ __launch_bounds__(32) void k_gamma_ctl(const float* __restrict__ mid,
                                                  float* P, float* __restrict__ ctl) {
  int sig = blockIdx.x, ch = threadIdx.x;
  GT g = gt_coeffs(P[P_BW1 + ch], P[P_CF + ch]);
  IIR4 sr = {0, 0, 0, 0}, si = {0, 0, 0, 0};
  const float* in = mid + (size_t)sig * NSAMP;
  float* out = ctl + ((size_t)sig * NCH + ch) * NSAMP;
  double ph = 0.0, ss = 0.0;
  for (int t = 0; t < NSAMP; ++t) {
    float cn, sn; __sincosf((float)ph, &sn, &cn);
    float x = in[t];
    float ur = iir4_step(sr, g, x * cn);
    float ui = iir4_step(si, g, x * sn);
    float e = g.gain * sqrtf(ur * ur + ui * ui);
    out[t] = e;
    ss += (double)e * (double)e;
    ph += g.w; if (ph >= TWO_PI_D) ph -= TWO_PI_D;
  }
  P[P_CTLRMS + sig * 32 + ch] = (float)sqrt(ss / (double)NSAMP);
}

__global__ __launch_bounds__(64) void k_bw(float* P, const int* level1p) {
  int t = threadIdx.x; int sig = t >> 5, ch = t & 31;
  float L1 = (float)(*level1p);
  float rms = P[P_CTLRMS + sig * 32 + ch];
  float cdb = 20.f * log10f(rms + SMALLV) + L1;
  float f = fminf(fmaxf((cdb - 50.f) / 50.f, 0.f), 1.f);
  float bmn = P[P_BWMIN + sig * 32 + ch], b1 = P[P_BW1 + ch];
  P[P_BW + sig * 32 + ch] = bmn + f * (b1 - bmn);
}

__global__ __launch_bounds__(32) void k_gamma_main(const float* __restrict__ mid, float* P,
                                                   float* __restrict__ env,
                                                   float* __restrict__ bm) {
  int sig = blockIdx.x, ch = threadIdx.x;
  GT g = gt_coeffs(P[P_BW + sig * 32 + ch], P[P_CF + ch]);
  IIR4 sr = {0, 0, 0, 0}, si = {0, 0, 0, 0};
  const float* in = mid + (size_t)sig * NSAMP;
  size_t base = ((size_t)sig * NCH + ch) * NSAMP;
  double ph = 0.0, ss = 0.0;
  for (int t = 0; t < NSAMP; ++t) {
    float cn, sn; __sincosf((float)ph, &sn, &cn);
    float x = in[t];
    float ur = iir4_step(sr, g, x * cn);
    float ui = iir4_step(si, g, x * sn);
    float e = g.gain * sqrtf(ur * ur + ui * ui);
    float b = g.gain * (ur * cn + ui * sn);
    env[base + t] = e; bm[base + t] = b;
    ss += (double)e * (double)e;
    ph += g.w; if (ph >= TWO_PI_D) ph -= TWO_PI_D;
  }
  P[P_ENVRMS + sig * 32 + ch] = (float)sqrt(ss / (double)NSAMP);
}

__global__ __launch_bounds__(32) void k_compress(const float* __restrict__ ctl,
                                                 float* __restrict__ env,
                                                 float* __restrict__ bm,
                                                 const float* P, const int* level1p) {
  int sig = blockIdx.x, ch = threadIdx.x;
  float L1 = (float)(*level1p);
  float attn = P[P_ATTN + sig * 32 + ch], knee = P[P_KNEE + sig * 32 + ch];
  float cr = P[P_CR + sig * 32 + ch];
  float slope = 1.f - 1.f / cr;
  float K = __tanf(3.14159265358979f * 800.f / 24000.f);
  float b0 = K / (1.f + K), b1 = b0, a1 = (K - 1.f) / (K + 1.f);
  float z = 0.f;
  size_t base = ((size_t)sig * NCH + ch) * NSAMP;
  for (int t = 0; t < NSAMP; ++t) {
    float c = fmaxf(ctl[base + t], SMALLV);
    float le = L1 + 20.f * log10f(c);
    le = fminf(fmaxf(le, knee), 100.f);
    float g0 = __expf((-attn - (le - knee) * slope) * 0.05f * 2.30258509299f);
    float g = b0 * g0 + z;
    z = b1 * g0 - a1 * g;
    env[base + t] *= g;
    bm[base + t] *= g;
  }
}

__global__ __launch_bounds__(256) void k_shift(const float* __restrict__ in,
                                               float* __restrict__ out,
                                               const int* __restrict__ delays) {
  int ch = blockIdx.y;
  int i = blockIdx.x * 256 + threadIdx.x;
  if (i >= NSAMP) return;
  int d = delays[ch];
  int j = i + d;
  out[(size_t)ch * NSAMP + i] = (j >= 0 && j < NSAMP) ? in[(size_t)ch * NSAMP + j] : 0.f;
}

__global__ __launch_bounds__(32) void k_sl_ihc(float* env0, float* bm0, float* env1,
                                               float* bm1, const float* P,
                                               const int* level1p) {
  int sig = blockIdx.x, ch = threadIdx.x;
  float* env = sig ? env1 : env0;
  float* bm  = sig ? bm1  : bm0;
  float L1 = (float)(*level1p);
  float ihc = P[P_IHC + sig * 32 + ch];
  const float r1 = 0.5f, r2 = 0.25f, r3 = 0.25f, T = 1.f / 24000.f;
  float c1 = 0.002f * (r1 + r2) / (r1 * r2), c2 = 0.060f / r3;
  float a11 = r1 + r2 + r1 * r2 * c1 / T, a12 = -r1;
  float a21 = -r3, a22 = r2 + r3 + r2 * r3 * c2 / T;
  float den = 1.f / (a11 * a22 - a21 * a12);
  float p1 = r1 * r2 * c1 / T, p2 = r2 * r3 * c2 / T;
  float v1 = 0.f, v2 = 0.f;
  size_t base = (size_t)ch * NSAMP;
  for (int t = 0; t < NSAMP; ++t) {
    float e = env[base + t], b = bm[base + t];
    float sl = fmaxf(L1 - ihc + 20.f * log10f(e + SMALLV), 0.f);
    float slb = sl * b / (e + SMALLV);
    float b1v = sl * r2 + p1 * v1;
    float b2v = p2 * v2;
    float nv1 = den * (a22 * b1v - a12 * b2v);
    float nv2 = den * (-a21 * b1v + a11 * b2v);
    v1 = nv1; v2 = nv2;
    float od = fmaxf((sl - nv1) / r1, 0.f);  // gain = r1+r2+r3 = 1
    float ob = slb * od / (sl + SMALLV);
    env[base + t] = od;
    bm[base + t] = ob;
  }
}

__global__ __launch_bounds__(256) void k_noise(float* bmx, float* bmy, const int* level1p) {
  size_t i = (size_t)blockIdx.x * 256 + threadIdx.x;
  if (i >= (size_t)NCH * NSAMP) return;
  float ng = __expf((-10.f - (float)(*level1p)) * 0.05f * 2.30258509299f);
  bmx[i] += ng * gaussf(0x1234u, (unsigned)i);
  bmy[i] += ng * gaussf(0x8765u, (unsigned)i);
}

__global__ __launch_bounds__(32) void k_gd(float* P) {
  int ch = threadIdx.x;
  GT g = gt_coeffs(P[P_BW + ch], P[P_CF + ch]);  // bw_x
  float tb = (g.a1 + 2.f * g.a5) / (1.f + g.a1 + g.a5);
  float ta = (-g.a1 - 2.f * g.a2 - 3.f * g.a3 - 4.f * g.a4) /
             (1.f - g.a1 - g.a2 - g.a3 - g.a4);
  int gd = (int)rintf(tb - ta);
  __shared__ int s[32];
  s[ch] = gd;
  __syncthreads();
  int mx = s[0];
  for (int i = 1; i < 32; ++i) mx = max(mx, s[i]);
  ((int*)P)[P_GD + ch] = mx - gd;
}

__global__ __launch_bounds__(256) void k_final(const float* s0, const float* s1,
                                               const float* s2, const float* s3,
                                               const float* P, float* out) {
  int ch = blockIdx.y, arr = blockIdx.z;
  int i = blockIdx.x * 256 + threadIdx.x;
  if (i >= NSAMP) return;
  const float* src = arr == 0 ? s0 : arr == 1 ? s1 : arr == 2 ? s2 : s3;
  int c = ((const int*)P)[P_GD + ch];
  int j = i - c;
  out[((size_t)arr * NCH + ch) * NSAMP + i] = (j >= 0) ? src[(size_t)ch * NSAMP + j] : 0.f;
}

__global__ __launch_bounds__(64) void k_sl(const float* P, const int* level1p, float* out) {
  int t = threadIdx.x;
  int sig = t >> 5, ch = t & 31;
  float L1 = (float)(*level1p);
  float avg = P[P_ENVRMS + sig * 32 + ch], cavg = P[P_CTLRMS + sig * 32 + ch];
  float attn = P[P_ATTN + sig * 32 + ch], knee = P[P_KNEE + sig * 32 + ch];
  float cr = P[P_CR + sig * 32 + ch], ihc = P[P_IHC + sig * 32 + ch];
  float cdb = fminf(fmaxf(L1 + 20.f * log10f(cavg + SMALLV), knee), 100.f);
  float gain = -attn - (cdb - knee) * (1.f - 1.f / cr);
  float sl = fmaxf(gain + L1 + 20.f * log10f(avg + SMALLV) - ihc, 0.f);
  out[(size_t)4 * NCH * NSAMP + sig * 32 + ch] = sl;
  if (t == 0) out[(size_t)4 * NCH * NSAMP + 64] = 24000.f;
}

// ================= launch =================
extern "C" void kernel_launch(void* const* d_in, const int* in_sizes, int n_in,
                              void* d_out, int out_size, void* d_ws, size_t ws_size,
                              hipStream_t stream) {
  (void)in_sizes; (void)n_in; (void)out_size; (void)ws_size;
  const float* ref    = (const float*)d_in[0];
  const float* proc   = (const float*)d_in[1];
  const float* hl     = (const float*)d_in[2];
  const int*   eq     = (const int*)d_in[5];
  const int*   level1 = (const int*)d_in[6];

  char* ws = (char*)d_ws;
  float* P = (float*)ws;
  size_t off = 4096;
  float* corr = (float*)(ws + off); off += (size_t)64 * NLBUF * 4;
  float* mid  = (float*)(ws + off); off += (size_t)2 * NSAMP * 4;
  float* ctl  = (float*)(ws + off); off += (size_t)2 * NCH * NSAMP * 4;
  float* env  = (float*)(ws + off); off += (size_t)2 * NCH * NSAMP * 4;
  float* bm   = (float*)(ws + off); off += (size_t)2 * NCH * NSAMP * 4;
  float* yal  = (float*)(ws + off); off += (size_t)2 * NCH * NSAMP * 4;
  _Float16* inh  = (_Float16*)(ws + off); off += (size_t)2 * PLEN * 2;   // ref/proc f16
  _Float16* envh = (_Float16*)(ws + off); off += (size_t)64 * PLEN * 2;  // xc|yc f16
  _Float16* bmh  = (_Float16*)(ws + off); off += (size_t)64 * PLEN * 2;  // xb|yb f16
  float* out  = (float*)d_out;
  int* dly0 = ((int*)P) + P_DELAY0;
  int* dlys = ((int*)P) + P_DELAYS;
  size_t half = (size_t)NCH * NSAMP;
  int pblk = (PLEN + 255) / 256;

  k_params<<<1, 32, 0, stream>>>(hl, eq, P);
  // input alignment via WMMA cross-correlation (padded f16 operands)
  k_tof16<<<dim3(pblk, 1), 256, 0, stream>>>(ref, inh);
  k_tof16<<<dim3(pblk, 1), 256, 0, stream>>>(proc, inh + PLEN);
  k_corr<<<dim3(NQBLK, 1), 32, 0, stream>>>(inh + PADE, inh + PLEN + PADE, 0, 0, corr);
  k_argmax<<<1, 256, 0, stream>>>(corr, dly0);
  k_midear<<<1, 32, 0, stream>>>(ref, proc, P, mid);
  k_gamma_ctl<<<2, 32, 0, stream>>>(mid, P, ctl);
  k_bw<<<1, 64, 0, stream>>>(P, level1);
  k_gamma_main<<<2, 32, 0, stream>>>(mid, P, env, bm);
  k_compress<<<2, 32, 0, stream>>>(ctl, env, bm, P, level1);
  // per-channel valign: 64 WMMA cross-correlations (xc/yc and xb/yb)
  k_tof16<<<dim3(pblk, 64), 256, 0, stream>>>(env, envh);
  k_tof16<<<dim3(pblk, 64), 256, 0, stream>>>(bm, bmh);
  k_corr<<<dim3(NQBLK, NCH), 32, 0, stream>>>(envh + PADE, envh + (size_t)32 * PLEN + PADE,
                                              PLEN, PLEN, corr);
  k_corr<<<dim3(NQBLK, NCH), 32, 0, stream>>>(bmh + PADE, bmh + (size_t)32 * PLEN + PADE,
                                              PLEN, PLEN, corr + (size_t)NCH * NLBUF);
  k_argmax<<<64, 256, 0, stream>>>(corr, dlys);
  k_shift<<<dim3((NSAMP + 255) / 256, NCH), 256, 0, stream>>>(env + half, yal, dlys);
  k_shift<<<dim3((NSAMP + 255) / 256, NCH), 256, 0, stream>>>(bm + half, yal + half,
                                                              dlys + NCH);
  k_sl_ihc<<<2, 32, 0, stream>>>(env, bm, yal, yal + half, P, level1);
  k_noise<<<(int)((half + 255) / 256), 256, 0, stream>>>(bm, yal + half, level1);
  k_gd<<<1, 32, 0, stream>>>(P);
  k_final<<<dim3((NSAMP + 255) / 256, NCH, 4), 256, 0, stream>>>(env, bm, yal, yal + half,
                                                                 P, out);
  k_sl<<<1, 64, 0, stream>>>(P, level1, out);
}
